// TreeLSTMCell_29386166239562
// MI455X (gfx1250) — matrix operand order repaired and checked by reading, not compiled, not run on hardware
//
#include <hip/hip_runtime.h>
#include <hip/hip_bf16.h>
#include <math.h>

#define N_NODES 262144
#define N_EDGES 262144
#define HD 128

typedef __bf16 v16bf_t __attribute__((ext_vector_type(16)));
typedef __bf16 v8bf_t  __attribute__((ext_vector_type(8)));
typedef float  v8f_t   __attribute__((ext_vector_type(8)));

// ---------------- stage 1: zero mailboxes + convert weights to bf16 ----------
__global__ void prep_zero_kernel(float4* __restrict__ dst, int n4) {
  int i = blockIdx.x * blockDim.x + threadIdx.x;
  if (i < n4) dst[i] = make_float4(0.f, 0.f, 0.f, 0.f);
}

// weight bf16 layout in ws: [W_iou 384*128][U_iou 384*128][W_f 128*128][U_f 128*128]
__global__ void prep_cvt_kernel(const float* __restrict__ Wiou,
                                const float* __restrict__ Uiou,
                                const float* __restrict__ Wf,
                                const float* __restrict__ Uf,
                                __bf16* __restrict__ out) {
  int t = blockIdx.x * blockDim.x + threadIdx.x;
  if (t < 49152)        out[t] = (__bf16)Wiou[t];
  else if (t < 98304)   out[t] = (__bf16)Uiou[t - 49152];
  else if (t < 114688)  out[t] = (__bf16)Wf[t - 98304];
  else if (t < 131072)  out[t] = (__bf16)Uf[t - 114688];
}

// ---------------- stage 2: scatter-add child h/c into parent mailboxes -------
__global__ void scatter_hc_kernel(const float* __restrict__ h,
                                  const float* __restrict__ c,
                                  const int* __restrict__ src,
                                  const int* __restrict__ dst,
                                  float* __restrict__ hsum,
                                  float* __restrict__ csum) {
  long long t = (long long)blockIdx.x * blockDim.x + threadIdx.x;
  int e   = (int)(t >> 7);
  int col = (int)(t & 127);
  if (e >= N_EDGES) return;
  int s = src[e];
  int d = dst[e];
  atomicAdd(&hsum[(long long)d * HD + col], h[(long long)s * HD + col]);
  atomicAdd(&csum[(long long)d * HD + col], c[(long long)s * HD + col]);
}

// ---------------- stage 3: fused bf16-WMMA GEMMs + gate pointwise ------------
__device__ __forceinline__ float fsigmoid(float x) {
  return 1.0f / (1.0f + __expf(-x));
}
__device__ __forceinline__ float ftanh(float x) {
  float e = __expf(-2.0f * x);
  return (1.0f - e) / (1.0f + e);
}

// A-fragment (16x32 bf16, MxK) from row-major LDS tile [16][128]:
//   lane 0-15 : M=lane,    halves 0-7 -> K=kb+0..7,   halves 8-15 -> K=kb+8..15
//   lane16-31 : M=lane-16, halves 0-7 -> K=kb+16..23, halves 8-15 -> K=kb+24..31
__device__ __forceinline__ v16bf_t load_a_frag(const __bf16* lds, int lane, int kb) {
  int row = lane & 15;
  int hi8 = (lane >> 4) << 4;                 // 0 or 16
  const __bf16* p = lds + row * HD + kb + hi8;
  v8bf_t lo = *(const v8bf_t*)(p);            // K = kb+hi8   .. +7   (16B aligned)
  v8bf_t hb = *(const v8bf_t*)(p + 8);        // K = kb+hi8+8 .. +15
  v16bf_t a;
#pragma unroll
  for (int j = 0; j < 8; j++) { a[j] = lo[j]; a[j + 8] = hb[j]; }
  return a;
}

// B-fragment (32x16 bf16, KxN) from row-major weight W[outdim][128] (bf16):
//   lane 0-15 : N=lane,    halves j -> K=kb+j
//   lane16-31 : N=lane-16, halves j -> K=kb+16+j
__device__ __forceinline__ v16bf_t load_b_frag(const __bf16* __restrict__ W,
                                               int colbase, int lane, int kb) {
  int n    = lane & 15;
  int hi16 = (lane >> 4) << 4;                // 0 or 16
  return *(const v16bf_t*)(W + (long long)(colbase + n) * HD + kb + hi16); // 32B aligned
}

#define WMMA_BF16(A, B, C) \
  __builtin_amdgcn_wmma_f32_16x16x32_bf16(false, (A), false, (B), (short)0, (C), false, false)

__global__ void __launch_bounds__(256)
fused_cell_kernel(const float* __restrict__ x,
                  const float* __restrict__ h,
                  const float* __restrict__ hsum,
                  const float* __restrict__ csum,
                  const __bf16* __restrict__ wts,
                  const float* __restrict__ b_iou,   // [384]
                  const float* __restrict__ Uf_b,    // [128]
                  const float* __restrict__ b_f,     // [128]
                  float* __restrict__ out) {
  __shared__ __bf16 sx[16 * HD];
  __shared__ __bf16 sh[16 * HD];
  __shared__ __bf16 ss[16 * HD];

  const int tid = threadIdx.x;
  const int rowbase = blockIdx.x * 16;

  // stage 16 rows of x / h / h_sum into LDS as bf16
  for (int i = tid; i < 16 * HD; i += 256) {
    long long g = (long long)rowbase * HD + i;
    sx[i] = (__bf16)x[g];
    sh[i] = (__bf16)h[g];
    ss[i] = (__bf16)hsum[g];
  }
  __syncthreads();

  const __bf16* Wiou_b = wts;
  const __bf16* Uiou_b = wts + 49152;
  const __bf16* Wf_b   = wts + 98304;
  const __bf16* Uf_w   = wts + 114688;

  const int w    = tid >> 5;          // wave id -> column tile
  const int lane = tid & 31;
  const int colbase = w << 4;
  const int c = colbase + (lane & 15);

  // bias-seeded accumulators (same col for all 8 C-rows of this lane)
  const float bF = Uf_b[c] + b_f[c];
  const float bI = b_iou[c];
  const float bO = b_iou[128 + c];
  const float bU = b_iou[256 + c];
  v8f_t accF, accI, accO, accU;
#pragma unroll
  for (int r = 0; r < 8; r++) { accF[r] = bF; accI[r] = bI; accO[r] = bO; accU[r] = bU; }

#pragma unroll
  for (int k = 0; k < 4; k++) {
    const int kb = k * 32;
    v16bf_t ax = load_a_frag(sx, lane, kb);
    v16bf_t ah = load_a_frag(sh, lane, kb);
    v16bf_t as = load_a_frag(ss, lane, kb);

    accF = WMMA_BF16(ax, load_b_frag(Wf_b,  colbase,        lane, kb), accF);
    accF = WMMA_BF16(ah, load_b_frag(Uf_w,  colbase,        lane, kb), accF);
    accI = WMMA_BF16(ax, load_b_frag(Wiou_b, colbase,       lane, kb), accI);
    accI = WMMA_BF16(as, load_b_frag(Uiou_b, colbase,       lane, kb), accI);
    accO = WMMA_BF16(ax, load_b_frag(Wiou_b, 128 + colbase, lane, kb), accO);
    accO = WMMA_BF16(as, load_b_frag(Uiou_b, 128 + colbase, lane, kb), accO);
    accU = WMMA_BF16(ax, load_b_frag(Wiou_b, 256 + colbase, lane, kb), accU);
    accU = WMMA_BF16(as, load_b_frag(Uiou_b, 256 + colbase, lane, kb), accU);
  }

  // gate pointwise + store; C layout: VGPR r -> M = r + (lane>=16 ? 8 : 0), N = lane&15
  const int rlo = rowbase + ((lane >> 4) << 3);
#pragma unroll
  for (int r = 0; r < 8; r++) {
    const int row = rlo + r;
    const float cs = csum[(long long)row * HD + c];
    const float f  = fsigmoid(accF[r]);
    const float cn = fsigmoid(accI[r]) * ftanh(accU[r]) + f * cs;
    const float hn = fsigmoid(accO[r]) * ftanh(cn);
    out[(long long)row * HD + c]                          = hn;  // h_new
    out[(long long)N_NODES * HD + (long long)row * HD + c] = cn; // c_new
  }
}

// -----------------------------------------------------------------------------
extern "C" void kernel_launch(void* const* d_in, const int* in_sizes, int n_in,
                              void* d_out, int out_size, void* d_ws, size_t ws_size,
                              hipStream_t stream) {
  const float* x    = (const float*)d_in[0];   // token_embeds [N,128]
  const float* h    = (const float*)d_in[1];   // h [N,128]
  const float* c    = (const float*)d_in[2];   // c [N,128]
  const int*   src  = (const int*)d_in[3];     // [E]
  const int*   dst  = (const int*)d_in[4];     // [E]
  const float* Wiou = (const float*)d_in[5];   // [384,128]
  const float* Uiou = (const float*)d_in[6];   // [384,128]
  const float* biou = (const float*)d_in[7];   // [1,384]
  const float* Ufw  = (const float*)d_in[8];   // [128,128]
  const float* Ufb  = (const float*)d_in[9];   // [128]
  const float* Wfw  = (const float*)d_in[10];  // [128,128]
  const float* bf   = (const float*)d_in[11];  // [1,128]

  // workspace layout: [hsum N*128 f32][csum N*128 f32][weights bf16 131072]
  float*  hsum = (float*)d_ws;
  float*  csum = hsum + (size_t)N_NODES * HD;
  __bf16* wts  = (__bf16*)(csum + (size_t)N_NODES * HD);

  const int n4 = (2 * N_NODES * HD) / 4;       // float4 zero-fill of mailboxes
  prep_zero_kernel<<<(n4 + 255) / 256, 256, 0, stream>>>((float4*)d_ws, n4);
  prep_cvt_kernel<<<(131072 + 255) / 256, 256, 0, stream>>>(Wiou, Uiou, Wfw, Ufw, wts);

  const long long nt = (long long)N_EDGES * HD;
  scatter_hc_kernel<<<(unsigned)(nt / 256), 256, 0, stream>>>(h, c, src, dst, hsum, csum);

  fused_cell_kernel<<<N_NODES / 16, 256, 0, stream>>>(x, h, hsum, csum, wts,
                                                      biou, Ufb, bf, (float*)d_out);
}
